// GAT_3152505995415
// MI455X (gfx1250) — matrix operand order repaired and checked by reading, not compiled, not run on hardware
//
#include <hip/hip_runtime.h>
#include <hip/hip_bf16.h>

typedef __attribute__((ext_vector_type(16))) _Float16 v16h;
typedef __attribute__((ext_vector_type(8)))  float    v8f;

#define N_HID 128   // HEADS*HID = 2*64

// ---------------------------------------------------------------------------
// WMMA GEMM: C[M x 128] = A[M x K] @ W[K x 128], f16 inputs, f32 accumulate.
// Block = 256 threads = 8 waves; wave w owns column tile [16w,16w+16).
// Each block processes RTILES row tiles (16*RTILES rows): W is staged to LDS
// once, each wave materializes its B fragments in registers once, then loops
// row tiles: 4x float4 A loads -> v_wmma_f32_16x16x32_f16 x KITER -> 8 stores.
// Requires nRows % (16*RTILES) == 0 (50000 = 625*80).
// ---------------------------------------------------------------------------
template <int K, int RTILES>
__global__ void gemm_wmma_kernel(const float* __restrict__ A,
                                 const float* __restrict__ W,
                                 float* __restrict__ C) {
    static_assert(K >= 128 || K <= 8, "small-K path assumes K fits lanes 0-15");
    constexpr int KP = 128;                        // padded K stride in LDS
    __shared__ __align__(32) _Float16 Bs[N_HID * KP];  // [n][k] transposed, 32 KB

    const int tid = threadIdx.x;
    for (int i = tid; i < N_HID * KP; i += 256) {
        const int n = i >> 7, k = i & 127;
        const float v = (k < K) ? W[k * N_HID + n] : 0.0f;
        Bs[n * KP + k] = (_Float16)v;
    }
    __syncthreads();

    const int wave = tid >> 5;
    const int lane = tid & 31;
    const int l15 = lane & 15;
    const int halfSel = lane >> 4;                 // 0: lanes 0-15, 1: lanes 16-31
    const int rowBase = blockIdx.x * (16 * RTILES);
    const int nBase = wave * 16;

    constexpr int KITER = (K + 31) / 32;

    // B fragments: k = 32t + 8*halfSel + e (e<8), 32t + 16 + 8*halfSel + (e-8) (e>=8).
    v16h bf[KITER];
    {
        const _Float16* bcol = &Bs[(nBase + l15) * KP + halfSel * 8];
#pragma unroll
        for (int t = 0; t < KITER; ++t) {
            const _Float16* bp = bcol + 32 * t;
#pragma unroll
            for (int e = 0; e < 16; ++e) bf[t][e] = bp[e + (e >> 3) * 8];
        }
    }

#pragma unroll 1
    for (int rt = 0; rt < RTILES; ++rt) {
        const int mrow = rowBase + rt * 16;
        v8f acc = {};
#pragma unroll
        for (int t = 0; t < KITER; ++t) {
            v16h a;
            if constexpr (K >= 128) {
                // unguarded vector loads: 8 + 8 consecutive floats
                const float* aptr = A + (size_t)(mrow + l15) * K + halfSel * 8;
                const float4 f0 = *(const float4*)(aptr + 32 * t);
                const float4 f1 = *(const float4*)(aptr + 32 * t + 4);
                const float4 f2 = *(const float4*)(aptr + 32 * t + 16);
                const float4 f3 = *(const float4*)(aptr + 32 * t + 20);
                a[0] = (_Float16)f0.x;  a[1] = (_Float16)f0.y;
                a[2] = (_Float16)f0.z;  a[3] = (_Float16)f0.w;
                a[4] = (_Float16)f1.x;  a[5] = (_Float16)f1.y;
                a[6] = (_Float16)f1.z;  a[7] = (_Float16)f1.w;
                a[8]  = (_Float16)f2.x; a[9]  = (_Float16)f2.y;
                a[10] = (_Float16)f2.z; a[11] = (_Float16)f2.w;
                a[12] = (_Float16)f3.x; a[13] = (_Float16)f3.y;
                a[14] = (_Float16)f3.z; a[15] = (_Float16)f3.w;
            } else {
                // K <= 8 (layer 0, K=7): only lanes 0-15 carry k = 0..K-1;
                // lanes 16-31 map to k >= 8 which is all zero-pad. Load the
                // row unconditionally (valid for every lane, no halfSel
                // offset -> no OOB) and mask branchlessly (v_cndmask).
                const float* arow = A + (size_t)(mrow + l15) * K;
#pragma unroll
                for (int j = 0; j < 8; ++j) {
                    const float v = (j < K) ? arow[j] : 0.0f;   // compile-time
                    a[j]     = (halfSel == 0) ? (_Float16)v : (_Float16)0.0f;
                    a[8 + j] = (_Float16)0.0f;                  // k >= 16
                }
            }
            acc = __builtin_amdgcn_wmma_f32_16x16x32_f16(
                false, a, false, bf[t], (short)0, acc, false, false);
        }
        // D layout: VGPR r -> row (r + 8*halfSel); unguarded contiguous stores
        float* cptr = C + (size_t)(mrow + halfSel * 8) * N_HID + nBase + l15;
#pragma unroll
        for (int r = 0; r < 8; ++r) cptr[r * N_HID] = acc[r];
    }
}

// s_i[n,h] = xp[n, h*64:+64] . att[h,0:64];  s_j: att[h,64:128]
__global__ void node_score_kernel(const float* __restrict__ xp,
                                  const float* __restrict__ att,
                                  float* __restrict__ s_i, float* __restrict__ s_j,
                                  int nNodes) {
    const int t = blockIdx.x * blockDim.x + threadIdx.x;
    const int n = t >> 1, h = t & 1;
    if (n >= nNodes) return;
    const float* xr = xp + (size_t)n * N_HID + h * 64;
    const float* ai = att + h * 192;
    const float* aj = att + h * 192 + 64;
    float si = 0.f, sj = 0.f;
#pragma unroll 4
    for (int c = 0; c < 64; ++c) { const float v = xr[c]; si += v * ai[c]; sj += v * aj[c]; }
    s_i[n * 2 + h] = si;
    s_j[n * 2 + h] = sj;
}

// we_ae[j,h] = w_edge[j, h*64:+64] . att[h,128:192]   (6x2)
__global__ void edge_proj_kernel(const float* __restrict__ w_edge,
                                 const float* __restrict__ att,
                                 float* __restrict__ we_ae) {
    const int t = threadIdx.x;
    if (t >= 12) return;
    const int j = t >> 1, h = t & 1;
    const float* ae = att + h * 192 + 128;
    float s = 0.f;
    for (int c = 0; c < 64; ++c) s += w_edge[j * N_HID + h * 64 + c] * ae[c];
    we_ae[j * 2 + h] = s;
}

// alpha[e,h]: leaky-relu then softmax over the 2 heads (axis=1 in reference)
__global__ void edge_alpha_kernel(const int* __restrict__ ei,
                                  const float* __restrict__ ea,
                                  const float* __restrict__ we_ae,
                                  const float* __restrict__ s_i,
                                  const float* __restrict__ s_j,
                                  float* __restrict__ alpha, int nEdges) {
    const int e = blockIdx.x * blockDim.x + threadIdx.x;
    if (e >= nEdges) return;
    const int src = ei[e];
    const int dst = ei[nEdges + e];
    float a0 = s_i[dst * 2 + 0] + s_j[src * 2 + 0];
    float a1 = s_i[dst * 2 + 1] + s_j[src * 2 + 1];
#pragma unroll
    for (int j = 0; j < 6; ++j) {
        const float v = ea[e * 6 + j];
        a0 += v * we_ae[j * 2 + 0];
        a1 += v * we_ae[j * 2 + 1];
    }
    a0 = (a0 > 0.f) ? a0 : 0.2f * a0;
    a1 = (a1 > 0.f) ? a1 : 0.2f * a1;
    const float mx = fmaxf(a0, a1);
    const float e0 = __expf(a0 - mx), e1 = __expf(a1 - mx);
    const float inv = 1.f / (e0 + e1);
    alpha[e * 2 + 0] = e0 * inv;
    alpha[e * 2 + 1] = e1 * inv;
}

// agg[n,c] = b[c]  (bias pre-seeded so the scatter only accumulates)
__global__ void init_bias_kernel(float* __restrict__ agg,
                                 const float* __restrict__ b, int total) {
    const int i = blockIdx.x * blockDim.x + threadIdx.x;
    if (i < total) agg[i] = b[i & (N_HID - 1)];
}

// One wave32 per edge; lane handles 4 consecutive channels (float4 load);
// lanes 0-15 -> head 0 channels, lanes 16-31 -> head 1 channels.
__global__ void scatter_kernel(const float* __restrict__ xp,
                               const float* __restrict__ alpha,
                               const int* __restrict__ ei,
                               float* __restrict__ agg, int nEdges) {
    const int gid = blockIdx.x * blockDim.x + threadIdx.x;
    const int e = gid >> 5;
    const int lane = gid & 31;
    if (e >= nEdges) return;
    const int src = ei[e];
    const int dst = ei[nEdges + e];
    const float al = alpha[e * 2 + (lane >> 4)];
    const float4 v = *(const float4*)(xp + (size_t)src * N_HID + lane * 4);
    float* dp = agg + (size_t)dst * N_HID + lane * 4;
    unsafeAtomicAdd(dp + 0, v.x * al);
    unsafeAtomicAdd(dp + 1, v.y * al);
    unsafeAtomicAdd(dp + 2, v.z * al);
    unsafeAtomicAdd(dp + 3, v.w * al);
}

__global__ void relu_kernel(float* __restrict__ h, int total) {
    const int i = blockIdx.x * blockDim.x + threadIdx.x;
    if (i < total) h[i] = fmaxf(h[i], 0.0f);
}

__global__ void zero_kernel(float* __restrict__ p, int total) {
    const int i = blockIdx.x * blockDim.x + threadIdx.x;
    if (i < total) p[i] = 0.0f;
}

// batch is sorted: run-length accumulate per block (64 nodes), flush on
// graph-id change -> ~2 atomic bursts per block instead of 64.
__global__ void pool_kernel(const float* __restrict__ h,
                            const int* __restrict__ batch,
                            float* __restrict__ sums, float* __restrict__ cnt,
                            int nNodes) {
    const int NB = 64;
    const int c = threadIdx.x;              // 128 threads = one channel each
    const int start = blockIdx.x * NB;
    if (start >= nNodes) return;
    const int end = min(start + NB, nNodes);
    int cur = batch[start];
    float acc = 0.f, cacc = 0.f;
    for (int n = start; n < end; ++n) {
        const int g = batch[n];
        if (g != cur) {
            unsafeAtomicAdd(&sums[cur * N_HID + c], acc);
            if (c == 0) unsafeAtomicAdd(&cnt[cur], cacc);
            acc = 0.f; cacc = 0.f; cur = g;
        }
        acc += h[(size_t)n * N_HID + c];
        cacc += 1.f;
    }
    unsafeAtomicAdd(&sums[cur * N_HID + c], acc);
    if (c == 0) unsafeAtomicAdd(&cnt[cur], cacc);
}

__global__ void head_kernel(const float* __restrict__ sums,
                            const float* __restrict__ cnt,
                            const float* __restrict__ w_out,
                            const float* __restrict__ b_out,
                            float* __restrict__ out, int nGraphs) {
    const int g = blockIdx.x * blockDim.x + threadIdx.x;
    if (g >= nGraphs) return;
    const float invc = 1.0f / fmaxf(cnt[g], 1.0f);
    float l0 = b_out[0], l1 = b_out[1];
    for (int k = 0; k < N_HID; ++k) {
        const float p = sums[g * N_HID + k] * invc;
        l0 += p * w_out[k * 2 + 0];
        l1 += p * w_out[k * 2 + 1];
    }
    const float mx = fmaxf(l0, l1);
    const float lse = mx + logf(__expf(l0 - mx) + __expf(l1 - mx));
    out[g * 2 + 0] = l0 - lse;
    out[g * 2 + 1] = l1 - lse;
}

extern "C" void kernel_launch(void* const* d_in, const int* in_sizes, int n_in,
                              void* d_out, int out_size, void* d_ws, size_t ws_size,
                              hipStream_t stream) {
    const float* x     = (const float*)d_in[0];   // (N,7)
    const int*   ei    = (const int*)  d_in[1];   // (2,E)
    const float* ea    = (const float*)d_in[2];   // (E,6)
    const int*   batch = (const int*)  d_in[3];   // (N,)
    const float* w_lin[3]  = {(const float*)d_in[4],  (const float*)d_in[8],  (const float*)d_in[12]};
    const float* w_edge[3] = {(const float*)d_in[5],  (const float*)d_in[9],  (const float*)d_in[13]};
    const float* att[3]    = {(const float*)d_in[6],  (const float*)d_in[10], (const float*)d_in[14]};
    const float* bias[3]   = {(const float*)d_in[7],  (const float*)d_in[11], (const float*)d_in[15]};
    const float* w_out = (const float*)d_in[16];
    const float* b_out = (const float*)d_in[17];
    float* out = (float*)d_out;

    const int N = in_sizes[0] / 7;      // 50000
    const int E = in_sizes[1] / 2;      // 640000
    const int G = out_size / 2;         // 64

    // Workspace carve (256B aligned)
    char* base = (char*)d_ws;
    size_t off = 0;
    auto carve = [&](size_t bytes) {
        void* p = base + off;
        off += (bytes + 255) & ~(size_t)255;
        return p;
    };
    float* H0   = (float*)carve((size_t)N * N_HID * 4);
    float* H1   = (float*)carve((size_t)N * N_HID * 4);
    float* XP   = (float*)carve((size_t)N * N_HID * 4);
    float* SI   = (float*)carve((size_t)N * 2 * 4);
    float* SJ   = (float*)carve((size_t)N * 2 * 4);
    float* AL   = (float*)carve((size_t)E * 2 * 4);
    float* WEAE = (float*)carve(256);
    float* SUMS = (float*)carve((size_t)G * N_HID * 4);
    float* CNT  = (float*)carve((size_t)G * 4);
    (void)ws_size; (void)n_in;

    constexpr int RT = 5;                       // 80 rows/block; 50000 = 625*80
    const int gemmBlocks = (N + 16 * RT - 1) / (16 * RT);
    const int totNC = N * N_HID;

    const float* layer_in[3] = {x, H0, H1};
    float*       layer_out[3] = {H0, H1, H0};

    for (int l = 0; l < 3; ++l) {
        // 1) xp = h @ w_lin  (WMMA)
        if (l == 0)
            gemm_wmma_kernel<7, RT><<<gemmBlocks, 256, 0, stream>>>(layer_in[l], w_lin[l], XP);
        else
            gemm_wmma_kernel<128, RT><<<gemmBlocks, 256, 0, stream>>>(layer_in[l], w_lin[l], XP);
        // 2) per-node attention scalars
        node_score_kernel<<<(2 * N + 255) / 256, 256, 0, stream>>>(XP, att[l], SI, SJ, N);
        // 3) fold w_edge @ a_e (6x2)
        edge_proj_kernel<<<1, 32, 0, stream>>>(w_edge[l], att[l], WEAE);
        // 4) per-edge alpha (leaky-relu + 2-head softmax)
        edge_alpha_kernel<<<(E + 255) / 256, 256, 0, stream>>>(ei, ea, WEAE, SI, SJ, AL, E);
        // 5) seed output with bias, 6) scatter-accumulate, 7) relu
        init_bias_kernel<<<(totNC + 255) / 256, 256, 0, stream>>>(layer_out[l], bias[l], totNC);
        scatter_kernel<<<(E * 32 + 255) / 256, 256, 0, stream>>>(XP, AL, ei, layer_out[l], E);
        relu_kernel<<<(totNC + 255) / 256, 256, 0, stream>>>(layer_out[l], totNC);
    }

    // Mean-pool per graph + linear head + log-softmax
    zero_kernel<<<(G * N_HID + 255) / 256, 256, 0, stream>>>(SUMS, G * N_HID);
    zero_kernel<<<1, 64, 0, stream>>>(CNT, G);
    pool_kernel<<<(N + 63) / 64, 128, 0, stream>>>(H0, batch, SUMS, CNT, N);
    head_kernel<<<1, 64, 0, stream>>>(SUMS, CNT, w_out, b_out, out, G);
}
